// SingleHeadAttn_12773232738995
// MI455X (gfx1250) — compile-verified, matrix operand order checked by reading, not run on hardware
//
#include <hip/hip_runtime.h>
#include <hip/hip_bf16.h>

typedef __bf16 bf16_t;
typedef bf16_t v16bf __attribute__((ext_vector_type(16)));
typedef bf16_t v8bf  __attribute__((ext_vector_type(8)));
typedef bf16_t v4bf  __attribute__((ext_vector_type(4)));
typedef float  v8f   __attribute__((ext_vector_type(8)));
typedef unsigned int u32;

#define BATCH 8
#define SEQ   2048
#define D_DIM 768
#define HD    64

#define USE_ASYNC_LDS 1

__device__ __forceinline__ bf16_t f2bf(float f) { return (bf16_t)f; }

__device__ __forceinline__ v16bf pack_frag(const bf16_t* p) {
  // p points at the 32-wide K-chunk base for this lane's half (kb already added)
  v8bf lo = *reinterpret_cast<const v8bf*>(p);
  v8bf hi = *reinterpret_cast<const v8bf*>(p + 16);
  return __builtin_shufflevector(lo, hi, 0, 1, 2, 3, 4, 5, 6, 7,
                                 8, 9, 10, 11, 12, 13, 14, 15);
}

__device__ __forceinline__ void async_copy_b128(const void* gsrc, void* ldst) {
#if USE_ASYNC_LDS
  u32 lds_addr = (u32)(uintptr_t)ldst;           // low 32 bits = LDS byte offset
  asm volatile("global_load_async_to_lds_b128 %0, %1, off"
               :: "v"(lds_addr), "v"(gsrc) : "memory");
#else
  *reinterpret_cast<uint4*>(ldst) = *reinterpret_cast<const uint4*>(gsrc);
#endif
}

// wait until at most N async ops remain outstanding (in-order completion)
__device__ __forceinline__ void async_wait_le4() {
#if USE_ASYNC_LDS
  asm volatile("s_wait_asynccnt 0x4" ::: "memory");
#endif
}
__device__ __forceinline__ void async_wait0() {
#if USE_ASYNC_LDS
  asm volatile("s_wait_asynccnt 0x0" ::: "memory");
#endif
}

// ---------------------------------------------------------------------------
// Kernel 1: fused QKV projection.  Block = 64 rows of (B*S), full 192 output
// cols (Q|K|V).  bf16 WMMA, fp32 accum, bias epilogue, bf16 output to ws.
// ---------------------------------------------------------------------------
#define XT_STRIDE 40   // 32 + 8 pad (bf16 elems), keeps 16B alignment
#define WT_STRIDE 40

__global__ __launch_bounds__(256)
void qkv_kernel(const float* __restrict__ X,
                const float* __restrict__ Wq, const float* __restrict__ bq,
                const float* __restrict__ Wk, const float* __restrict__ bk,
                const float* __restrict__ Wv, const float* __restrict__ bv,
                bf16_t* __restrict__ Qb, bf16_t* __restrict__ Kb,
                bf16_t* __restrict__ Vb) {
  const int row0 = blockIdx.x * 64;          // row in flattened [B*S]
  const int tid  = threadIdx.x;
  const int lane = tid & 31;
  const int wave = tid >> 5;                 // 0..7
  const int l15  = lane & 15;
  const int kb   = (lane < 16) ? 0 : 8;      // K sub-base per 16-lane group

  __shared__ bf16_t Xt[64 * XT_STRIDE];      // [row][d]  (row-major)
  __shared__ bf16_t WtT[192 * WT_STRIDE];    // [n][d]    (transposed)

  v8f zero = {0.f, 0.f, 0.f, 0.f, 0.f, 0.f, 0.f, 0.f};
  v8f acc[6];
#pragma unroll
  for (int t = 0; t < 6; ++t) acc[t] = zero;

  for (int kc = 0; kc < D_DIM / 32; ++kc) {
    const int d0 = kc * 32;
    __syncthreads();
    // ---- cooperative load: X chunk [64 x 32] fp32 -> bf16 LDS ----
    for (int c = tid; c < 512; c += 256) {           // 512 float4 chunks
      int r  = c >> 3;
      int dd = (c & 7) * 4;
      const float4 v = *reinterpret_cast<const float4*>(
          X + (size_t)(row0 + r) * D_DIM + d0 + dd);
      v4bf t;
      t[0] = f2bf(v.x); t[1] = f2bf(v.y); t[2] = f2bf(v.z); t[3] = f2bf(v.w);
      *reinterpret_cast<v4bf*>(&Xt[r * XT_STRIDE + dd]) = t;
    }
    // ---- W chunk [32 x 64] x3, transposed into WtT[n][d] ----
    for (int c = tid; c < 1536; c += 256) {
      int m  = c / 512;                               // which matrix
      int cc = c - m * 512;
      int dd = cc >> 4;                               // d row (0..31)
      int nn = (cc & 15) * 4;                         // col base
      const float* Wm = (m == 0) ? Wq : (m == 1) ? Wk : Wv;
      const float4 v = *reinterpret_cast<const float4*>(
          Wm + (size_t)(d0 + dd) * HD + nn);
      int nbase = m * 64 + nn;
      WtT[(nbase + 0) * WT_STRIDE + dd] = f2bf(v.x);
      WtT[(nbase + 1) * WT_STRIDE + dd] = f2bf(v.y);
      WtT[(nbase + 2) * WT_STRIDE + dd] = f2bf(v.z);
      WtT[(nbase + 3) * WT_STRIDE + dd] = f2bf(v.w);
    }
    __syncthreads();

#pragma unroll
    for (int t = 0; t < 6; ++t) {
      int job = wave + t * 8;                 // 48 tile-jobs
      int jm = job & 3;                       // m-tile (0..3)
      int jn = job >> 2;                      // n-tile (0..11)
      v16bf A = pack_frag(&Xt[(jm * 16 + l15) * XT_STRIDE + kb]);
      v16bf B = pack_frag(&WtT[(jn * 16 + l15) * WT_STRIDE + kb]);
      acc[t] = __builtin_amdgcn_wmma_f32_16x16x32_bf16(
          false, A, false, B, (short)0, acc[t], false, false);
    }
  }

  // ---- epilogue: + bias, -> bf16 ----
#pragma unroll
  for (int t = 0; t < 6; ++t) {
    int job = wave + t * 8;
    int jm = job & 3;
    int jn = job >> 2;
    int mat = jn >> 2;                         // 0=Q 1=K 2=V
    int col = (jn & 3) * 16 + l15;
    const float* bp = (mat == 0) ? bq : (mat == 1) ? bk : bv;
    bf16_t* op = (mat == 0) ? Qb : (mat == 1) ? Kb : Vb;
    float bias = bp[col];
    int rbase = row0 + jm * 16 + ((lane < 16) ? 0 : 8);
#pragma unroll
    for (int r = 0; r < 8; ++r)
      op[(size_t)(rbase + r) * HD + col] = f2bf(acc[t][r] + bias);
  }
}

// ---------------------------------------------------------------------------
// Kernel 2: causal flash attention.  Block = (batch, 64 q rows), 4 waves,
// wave = 16 q rows.  Online softmax within a 16-lane group.
//   K tiles: double-buffered async-DMA copies (ASYNCcnt ping-pong; the copy
//            for tile j+1 stays in flight across tile j's WMMA compute).
//   V tiles: register-pipelined (next tile's global loads issued alongside
//            the async K issue, scattered/transposed into LDS next iter).
// ---------------------------------------------------------------------------
#define KT_STRIDE 72   // 64 + 8 pad (bf16), preserves 16B alignment (144B)

__global__ __launch_bounds__(128)
void attn_kernel(const bf16_t* __restrict__ Qb, const bf16_t* __restrict__ Kb,
                 const bf16_t* __restrict__ Vb, float* __restrict__ Out) {
  const int b    = blockIdx.y;
  const int qt   = blockIdx.x;               // 64-row q tile
  const int tid  = threadIdx.x;
  const int lane = tid & 31;
  const int wave = tid >> 5;                 // 0..3
  const int l15  = lane & 15;
  const int kb   = (lane < 16) ? 0 : 8;

  __shared__ bf16_t Kt[2][64 * KT_STRIDE];   // [buf][key][d]
  __shared__ bf16_t VtT[64 * KT_STRIDE];     // [d][key]  (transposed)
  __shared__ bf16_t Pt[4 * 16 * KT_STRIDE];  // wave-private [row][key]

  const size_t baseBS = (size_t)b * SEQ;
  const int q0 = qt * 64 + wave * 16;        // this wave's first q row

  // Q A-fragments (held for whole kernel)
  v16bf qa[2];
  {
    const bf16_t* qrow = Qb + (baseBS + q0 + l15) * HD;
#pragma unroll
    for (int dc = 0; dc < 2; ++dc)
      qa[dc] = pack_frag(qrow + dc * 32 + kb);
  }

  v8f zero = {0.f, 0.f, 0.f, 0.f, 0.f, 0.f, 0.f, 0.f};
  v8f o[4];
#pragma unroll
  for (int n = 0; n < 4; ++n) o[n] = zero;
  float mrun[8], lrun[8];
#pragma unroll
  for (int r = 0; r < 8; ++r) { mrun[r] = -INFINITY; lrun[r] = 0.f; }

  const float inv_scale = 0.03608439182435161f;   // 1/sqrt(768)
  const int qrow_g = q0 + ((lane < 16) ? 0 : 8);  // + r gives global q row
  bf16_t* pw = &Pt[wave * 16 * KT_STRIDE];

  // ---- prologue: async K tile 0, V tile 0 into registers ----
#pragma unroll
  for (int i = 0; i < 4; ++i) {
    int c = tid + i * 128;
    int r = c >> 3, dd = (c & 7) * 8;
    async_copy_b128(Kb + (baseBS + r) * HD + dd, &Kt[0][r * KT_STRIDE + dd]);
  }
  v8bf vreg[4];
#pragma unroll
  for (int i = 0; i < 4; ++i) {
    int c = tid + i * 128;
    int r = c >> 3, dd = (c & 7) * 8;
    vreg[i] = *reinterpret_cast<const v8bf*>(Vb + (baseBS + r) * HD + dd);
  }

  for (int j = 0; j <= qt; ++j) {            // causal: skip masked key tiles
    const int cur = j & 1;
    __syncthreads();                         // all reads of prev tile done
    // ---- scatter V tile j from registers into VtT[d][key] ----
#pragma unroll
    for (int i = 0; i < 4; ++i) {
      int c = tid + i * 128;
      int r = c >> 3, dd = (c & 7) * 8;
#pragma unroll
      for (int e = 0; e < 8; ++e) VtT[(dd + e) * KT_STRIDE + r] = vreg[i][e];
    }
    if (j < qt) {
      // issue async K tile j+1 into the other buffer (stays in flight)
#pragma unroll
      for (int i = 0; i < 4; ++i) {
        int c = tid + i * 128;
        int r = c >> 3, dd = (c & 7) * 8;
        async_copy_b128(Kb + (baseBS + (j + 1) * 64 + r) * HD + dd,
                        &Kt[cur ^ 1][r * KT_STRIDE + dd]);
      }
      // prefetch V tile j+1 into registers (consumed after next barrier)
#pragma unroll
      for (int i = 0; i < 4; ++i) {
        int c = tid + i * 128;
        int r = c >> 3, dd = (c & 7) * 8;
        vreg[i] = *reinterpret_cast<const v8bf*>(
            Vb + (baseBS + (j + 1) * 64 + r) * HD + dd);
      }
      async_wait_le4();                      // tile j's 4 chunks done (in-order)
    } else {
      async_wait0();
    }
    __syncthreads();

    // ---- S = Q * K^T  (4 n-tiles x 2 K-chunks) ----
    const bf16_t* kt = Kt[cur];
    v8f s[4];
#pragma unroll
    for (int n = 0; n < 4; ++n) s[n] = zero;
#pragma unroll
    for (int dc = 0; dc < 2; ++dc) {
#pragma unroll
      for (int n = 0; n < 4; ++n) {
        v16bf B = pack_frag(&kt[(n * 16 + l15) * KT_STRIDE + dc * 32 + kb]);
        s[n] = __builtin_amdgcn_wmma_f32_16x16x32_bf16(
            false, qa[dc], false, B, (short)0, s[n], false, false);
      }
    }

    // ---- scale + causal mask + row max ----
    float tmax[8];
#pragma unroll
    for (int r = 0; r < 8; ++r) tmax[r] = -INFINITY;
#pragma unroll
    for (int n = 0; n < 4; ++n) {
      int kcol = j * 64 + n * 16 + l15;
#pragma unroll
      for (int r = 0; r < 8; ++r) {
        float v = s[n][r] * inv_scale;
        v = (kcol <= qrow_g + r) ? v : -INFINITY;
        s[n][r] = v;
        tmax[r] = fmaxf(tmax[r], v);
      }
    }
#pragma unroll
    for (int off = 1; off < 16; off <<= 1)
#pragma unroll
      for (int r = 0; r < 8; ++r)
        tmax[r] = fmaxf(tmax[r], __shfl_xor(tmax[r], off));

    // ---- online softmax update ----
    float corr[8];
#pragma unroll
    for (int r = 0; r < 8; ++r) {
      float mnew = fmaxf(mrun[r], tmax[r]);
      corr[r] = __expf(mrun[r] - mnew);
      mrun[r] = mnew;
      lrun[r] *= corr[r];
    }
#pragma unroll
    for (int n = 0; n < 4; ++n)
#pragma unroll
      for (int r = 0; r < 8; ++r) o[n][r] *= corr[r];

    // ---- P = exp(S - m), row-sum, spill to LDS in [row][key] ----
    float tsum[8];
#pragma unroll
    for (int r = 0; r < 8; ++r) tsum[r] = 0.f;
#pragma unroll
    for (int n = 0; n < 4; ++n) {
#pragma unroll
      for (int r = 0; r < 8; ++r) {
        float p = __expf(s[n][r] - mrun[r]);
        tsum[r] += p;
        pw[(r + ((lane < 16) ? 0 : 8)) * KT_STRIDE + n * 16 + l15] = f2bf(p);
      }
    }
#pragma unroll
    for (int off = 1; off < 16; off <<= 1)
#pragma unroll
      for (int r = 0; r < 8; ++r) tsum[r] += __shfl_xor(tsum[r], off);
#pragma unroll
    for (int r = 0; r < 8; ++r) lrun[r] += tsum[r];

    // ---- O += P * V ----
#pragma unroll
    for (int dc = 0; dc < 2; ++dc) {
      v16bf A = pack_frag(pw + l15 * KT_STRIDE + dc * 32 + kb);
#pragma unroll
      for (int n = 0; n < 4; ++n) {
        v16bf B = pack_frag(&VtT[(n * 16 + l15) * KT_STRIDE + dc * 32 + kb]);
        o[n] = __builtin_amdgcn_wmma_f32_16x16x32_bf16(
            false, A, false, B, (short)0, o[n], false, false);
      }
    }
  }

  // ---- normalize and store fp32 ----
#pragma unroll
  for (int n = 0; n < 4; ++n) {
#pragma unroll
    for (int r = 0; r < 8; ++r) {
      int row = qrow_g + r;
      Out[(baseBS + row) * HD + n * 16 + l15] = o[n][r] / lrun[r];
    }
  }
}

// ---------------------------------------------------------------------------
extern "C" void kernel_launch(void* const* d_in, const int* in_sizes, int n_in,
                              void* d_out, int out_size, void* d_ws,
                              size_t ws_size, hipStream_t stream) {
  const float* X  = (const float*)d_in[0];
  const float* Wq = (const float*)d_in[1];
  const float* bq = (const float*)d_in[2];
  const float* Wk = (const float*)d_in[3];
  const float* bk = (const float*)d_in[4];
  const float* Wv = (const float*)d_in[5];
  const float* bv = (const float*)d_in[6];
  float* Out = (float*)d_out;

  const size_t per = (size_t)BATCH * SEQ * HD;   // 1M elems each
  bf16_t* Qb = (bf16_t*)d_ws;
  bf16_t* Kb = Qb + per;
  bf16_t* Vb = Kb + per;

  qkv_kernel<<<dim3((BATCH * SEQ) / 64), 256, 0, stream>>>(
      X, Wq, bq, Wk, bk, Wv, bv, Qb, Kb, Vb);
  attn_kernel<<<dim3(SEQ / 64, BATCH), 128, 0, stream>>>(Qb, Kb, Vb, Out);
}
